// MoEMLP_74131135529703
// MI455X (gfx1250) — compile-verified, matrix operand order checked by reading, not run on hardware
//
#include <hip/hip_runtime.h>
#include <hip/hip_bf16.h>

// ---------------------------------------------------------------------------
// MoE SwiGLU MLP for MI455X (gfx1250): sparse top-2 routing + bf16 WMMA,
// with global_prefetch of the next weight chunk to hide HBM/L2 latency.
// ---------------------------------------------------------------------------

#define HIDDEN 1024
#define INTER  4096
#define NEXP   8
#define TOKENS 2048          // BATCH * SEQ
#define TILE_M 16            // tokens per expert tile (one WMMA M)
#define NTHREADS 256
#define FCHUNK 128           // F columns per outer iteration (16 per wave)
#define MAX_TILES 128        // worst case TOKENS/TILE_M per expert

typedef __attribute__((ext_vector_type(16))) __bf16 v16bf;
typedef __attribute__((ext_vector_type(8)))  float  v8f;

// float -> bf16 bits, round-to-nearest-even
__device__ __forceinline__ unsigned short f2bf(float f) {
    unsigned int u = __float_as_uint(f);
    unsigned int r = u + 0x7FFFu + ((u >> 16) & 1u);
    return (unsigned short)(r >> 16);
}

// 16-bit WMMA operand layout: k (0..31) -> slot s (0..15) and lane half.
// Lanes 0-15 hold K in {0..7} U {16..23}; lanes 16-31 hold {8..15} U {24..31}.
__device__ __forceinline__ int kslot(int kin, int& khalf) {
    khalf     = (kin >> 3) & 1;
    int group = kin >> 4;
    int r     = kin & 7;
    int v     = (group << 2) + (r >> 1);
    return v * 2 + (r & 1);
}

__device__ __forceinline__ v16bf ldv16(const unsigned short* p) {
    return *reinterpret_cast<const v16bf*>(p);
}

// ---------------------------------------------------------------------------
// Kernel 1: router.  One wave32 per token.
// ---------------------------------------------------------------------------
__global__ __launch_bounds__(NTHREADS, 1)
void moe_router_kernel(const float* __restrict__ x,
                       const float* __restrict__ gate_w,
                       int* __restrict__ cnt,
                       int* __restrict__ ids,
                       float* __restrict__ wts) {
    const int lane = threadIdx.x & 31;
    const int wv   = threadIdx.x >> 5;
    const int t    = blockIdx.x * (NTHREADS / 32) + wv;
    if (t >= TOKENS) return;

    float acc[NEXP];
#pragma unroll
    for (int e = 0; e < NEXP; ++e) acc[e] = 0.f;

    const float* xr = x + (size_t)t * HIDDEN;
    for (int h = lane; h < HIDDEN; h += 32) {
        float xv = xr[h];
        const float4* gw4 = reinterpret_cast<const float4*>(gate_w + h * NEXP);
        float4 a = gw4[0], b = gw4[1];
        acc[0] += xv * a.x; acc[1] += xv * a.y; acc[2] += xv * a.z; acc[3] += xv * a.w;
        acc[4] += xv * b.x; acc[5] += xv * b.y; acc[6] += xv * b.z; acc[7] += xv * b.w;
    }
#pragma unroll
    for (int off = 16; off > 0; off >>= 1)
#pragma unroll
        for (int e = 0; e < NEXP; ++e)
            acc[e] += __shfl_xor(acc[e], off, 32);

    if (lane == 0) {
        float mx = acc[0];
#pragma unroll
        for (int e = 1; e < NEXP; ++e) mx = fmaxf(mx, acc[e]);
        float p[NEXP], sum = 0.f;
#pragma unroll
        for (int e = 0; e < NEXP; ++e) { p[e] = __expf(acc[e] - mx); sum += p[e]; }
        float inv = 1.f / sum;
#pragma unroll
        for (int e = 0; e < NEXP; ++e) p[e] *= inv;

        int i1 = 0;
#pragma unroll
        for (int e = 1; e < NEXP; ++e) if (p[e] > p[i1]) i1 = e;
        int i2 = (i1 == 0) ? 1 : 0;
#pragma unroll
        for (int e = 0; e < NEXP; ++e) if (e != i1 && p[e] > p[i2]) i2 = e;

        // re-softmax over the two top PROBS (matches the reference)
        float w1 = 1.f / (1.f + __expf(p[i2] - p[i1]));
        float w2 = 1.f - w1;

        int s1 = atomicAdd(&cnt[i1], 1);
        ids[i1 * TOKENS + s1] = t;  wts[i1 * TOKENS + s1] = w1;
        int s2 = atomicAdd(&cnt[i2], 1);
        ids[i2 * TOKENS + s2] = t;  wts[i2 * TOKENS + s2] = w2;
    }
}

// ---------------------------------------------------------------------------
// Kernel 2: fused expert SwiGLU MLP (bf16 WMMA), grid = NEXP * MAX_TILES.
// Dynamic LDS layout (ushort units):
//   xs  [32 kblk][32 lane][16]          = 16384  (x tile, A-layout, K=HIDDEN)
//   hs  [ 4 kblk][32 lane][16]          =  2048  (silu(g)*u, A-layout, K=128)
//   wst 32768 ushorts (64 KB), reused:
//     phase 1: wg [4 kbi][8 ftile][32][16] = 16384 ; wu same at +16384
//     phase 2: wd [64 ntile][32][16]       = 32768
// ---------------------------------------------------------------------------
#define XS_OFF  0
#define HS_OFF  16384
#define WS_OFF  18432
#define SMEM_USHORTS (WS_OFF + 32768)

__global__ __launch_bounds__(NTHREADS, 1)
void moe_expert_kernel(const float* __restrict__ x,
                       const float* __restrict__ w_gate,
                       const float* __restrict__ w_up,
                       const float* __restrict__ w_down,
                       const int*   __restrict__ ids,
                       const float* __restrict__ wts,
                       const int*   __restrict__ cnt,
                       float* __restrict__ out) {
    extern __shared__ unsigned short smem[];
    unsigned short* xs  = smem + XS_OFF;
    unsigned short* hs  = smem + HS_OFF;
    unsigned short* wgs = smem + WS_OFF;            // phase 1: gate (4x8 tiles)
    unsigned short* wus = smem + WS_OFF + 16384;    // phase 1: up   (4x8 tiles)
    unsigned short* wds = smem + WS_OFF;            // phase 2: down (64 tiles)

    __shared__ int   stok[TILE_M];
    __shared__ float swt[TILE_M];

    const int tid  = threadIdx.x;
    const int lane = tid & 31;
    const int wv   = tid >> 5;
    const int e    = blockIdx.x >> 7;        // / MAX_TILES
    const int tile = blockIdx.x & (MAX_TILES - 1);

    const int cntE = cnt[e];
    const int nTok = min(TILE_M, cntE - tile * TILE_M);
    if (nTok <= 0) return;

    if (tid < TILE_M) {
        int r = tile * TILE_M + tid;
        bool in = (r < cntE);
        stok[tid] = in ? ids[e * TOKENS + r] : 0;
        swt[tid]  = in ? wts[e * TOKENS + r] : 0.f;
    }
    __syncthreads();

    // ---- stage x tile into LDS in WMMA A layout, fp32 -> bf16 -------------
    for (int i = tid; i < TILE_M * HIDDEN; i += NTHREADS) {
        int m = i >> 10;            // token row 0..15
        int h = i & (HIDDEN - 1);   // contiguous across threads -> coalesced
        float val = x[(size_t)stok[m] * HIDDEN + h];
        int kb = h >> 5, kin = h & 31, kh;
        int s = kslot(kin, kh);
        xs[((kb * 32) + (kh * 16 + m)) * 16 + s] = f2bf(val);
    }
    __syncthreads();

    const float* wGe = w_gate + (size_t)e * HIDDEN * INTER;
    const float* wUe = w_up   + (size_t)e * HIDDEN * INTER;
    const float* wDe = w_down + (size_t)e * INTER * HIDDEN;

    // down-proj accumulators: wave wv owns H columns [wv*128, wv*128+128)
    v8f accd[8];
#pragma unroll
    for (int i = 0; i < 8; ++i) accd[i] = {};

    const int khalfL = lane >> 4;
    const int nl     = lane & 15;

    for (int fc = 0; fc < INTER / FCHUNK; ++fc) {
        const int fBase = fc * FCHUNK;
        v8f accg = {}, accu = {};

        // ---- phase 1: g = x@Wg, u = x@Wu over K = HIDDEN ------------------
        // Stage K=128 x F=128 of both weight matrices per barrier pair,
        // then run 4 K-steps (8 WMMAs per wave) between barriers.
        for (int kbo = 0; kbo < HIDDEN / 128; ++kbo) {
            const int hBase = kbo * 128;
            for (int i = tid; i < 128 * FCHUNK; i += NTHREADS) {
                int kin128 = i >> 7;          // 0..127
                int fcol   = i & (FCHUNK - 1);
                int kbi = kin128 >> 5, kin = kin128 & 31, kh;
                int ftile = fcol >> 4, n = fcol & 15;
                int s = kslot(kin, kh);
                int dst = (((kbi * 8) + ftile) * 32 + (kh * 16 + n)) * 16 + s;
                size_t src = (size_t)(hBase + kin128) * INTER + fBase + fcol;
                wgs[dst] = f2bf(wGe[src]);
                wus[dst] = f2bf(wUe[src]);
            }

            // speculative prefetch of the NEXT 128x128 chunk (one 128B line
            // per prefetch, 512 lines per matrix spread over 256 threads)
            {
                int nkbo = kbo + 1, nfB = fBase;
                if (nkbo == HIDDEN / 128) { nkbo = 0; nfB = fBase + FCHUNK; }
                if (nfB < INTER) {
#pragma unroll
                    for (int it = 0; it < 2; ++it) {
                        int idx = tid + it * NTHREADS;     // 0..511
                        int r = idx >> 2, l = idx & 3;     // row, line-in-row
                        size_t off = (size_t)(nkbo * 128 + r) * INTER + nfB + l * 32;
                        __builtin_prefetch(wGe + off, 0, 0);
                        __builtin_prefetch(wUe + off, 0, 0);
                    }
                }
            }
            __syncthreads();

#pragma unroll
            for (int kbi = 0; kbi < 4; ++kbi) {
                v16bf a  = ldv16(&xs[((kbo * 4 + kbi) * 32 + lane) * 16]);
                v16bf bg = ldv16(&wgs[((kbi * 8 + wv) * 32 + lane) * 16]);
                v16bf bu = ldv16(&wus[((kbi * 8 + wv) * 32 + lane) * 16]);
                accg = __builtin_amdgcn_wmma_f32_16x16x32_bf16(
                    false, a, false, bg, (short)0, accg, false, false);
                accu = __builtin_amdgcn_wmma_f32_16x16x32_bf16(
                    false, a, false, bu, (short)0, accu, false, false);
            }
            __syncthreads();
        }

        // ---- silu(g) * u -> hs (A layout, K = FCHUNK) ---------------------
#pragma unroll
        for (int j = 0; j < 8; ++j) {
            int m   = j + 8 * khalfL;
            float g = accg[j], u = accu[j];
            float hv = (g / (1.f + __expf(-g))) * u;
            int k128 = wv * 16 + nl;        // f-local index within chunk
            int kb2 = k128 >> 5, kin = k128 & 31, kh;
            int s = kslot(kin, kh);
            hs[((kb2 * 32) + (kh * 16 + m)) * 16 + s] = f2bf(hv);
        }
        __syncthreads();

        // ---- phase 2: out += h @ Wdown, K = FCHUNK ------------------------
        for (int ks = 0; ks < FCHUNK / 32; ++ks) {
            // stage Wdown [32 x 1024] slice into B layout (all 64 n-tiles)
            for (int i = tid; i < 32 * HIDDEN; i += NTHREADS) {
                int kin  = i >> 10;
                int hcol = i & (HIDDEN - 1);
                int ntile = hcol >> 4, n = hcol & 15, kh;
                int s = kslot(kin, kh);
                wds[((ntile * 32) + (kh * 16 + n)) * 16 + s] =
                    f2bf(wDe[(size_t)(fBase + ks * 32 + kin) * HIDDEN + hcol]);
            }

            // speculative prefetch of the NEXT Wdown 32x1024 slice
            {
                int nf = fBase + (ks + 1) * 32;   // next slice's first F row
                if (nf < INTER) {
#pragma unroll
                    for (int it = 0; it < 4; ++it) {
                        int idx = tid + it * NTHREADS;     // 0..1023 lines
                        int r = idx >> 5, l = idx & 31;
                        __builtin_prefetch(
                            wDe + (size_t)(nf + r) * HIDDEN + l * 32, 0, 0);
                    }
                }
            }
            __syncthreads();

            v16bf a2 = ldv16(&hs[(ks * 32 + lane) * 16]);
#pragma unroll
            for (int ntl = 0; ntl < 8; ++ntl) {
                int nt = wv * 8 + ntl;
                v16bf b = ldv16(&wds[(nt * 32 + lane) * 16]);
                accd[ntl] = __builtin_amdgcn_wmma_f32_16x16x32_bf16(
                    false, a2, false, b, (short)0, accd[ntl], false, false);
            }
            __syncthreads();
        }
    }

    // ---- weighted combine into output (f32 global atomics) ----------------
#pragma unroll
    for (int ntl = 0; ntl < 8; ++ntl) {
#pragma unroll
        for (int j = 0; j < 8; ++j) {
            int m = j + 8 * khalfL;
            if (m < nTok) {
                int hcol = wv * 128 + ntl * 16 + nl;
                atomicAdd(&out[(size_t)stok[m] * HIDDEN + hcol],
                          accd[ntl][j] * swt[m]);
            }
        }
    }
}

// ---------------------------------------------------------------------------
// Host launcher
// ---------------------------------------------------------------------------
extern "C" void kernel_launch(void* const* d_in, const int* in_sizes, int n_in,
                              void* d_out, int out_size, void* d_ws, size_t ws_size,
                              hipStream_t stream) {
    const float* x      = (const float*)d_in[0];
    const float* gate_w = (const float*)d_in[1];
    const float* w_gate = (const float*)d_in[2];
    const float* w_up   = (const float*)d_in[3];
    const float* w_down = (const float*)d_in[4];
    float* out = (float*)d_out;

    // workspace layout
    int*   cnt = (int*)d_ws;                       // 16 ints (8 used)
    int*   ids = cnt + 16;                         // NEXP * TOKENS ints
    float* wts = (float*)(ids + NEXP * TOKENS);    // NEXP * TOKENS floats

    hipMemsetAsync(cnt, 0, 16 * sizeof(int), stream);
    hipMemsetAsync(out, 0, (size_t)TOKENS * HIDDEN * sizeof(float), stream);

    moe_router_kernel<<<TOKENS / (NTHREADS / 32), NTHREADS, 0, stream>>>(
        x, gate_w, cnt, ids, wts);

    const int smem_bytes = SMEM_USHORTS * (int)sizeof(unsigned short); // 102400
    hipFuncSetAttribute((const void*)moe_expert_kernel,
                        hipFuncAttributeMaxDynamicSharedMemorySize, smem_bytes);

    moe_expert_kernel<<<dim3(NEXP * MAX_TILES), NTHREADS, smem_bytes, stream>>>(
        x, w_gate, w_up, w_down, ids, wts, cnt, out);
}
